// AttentionFusion_16965120820095
// MI455X (gfx1250) — matrix operand order repaired and checked by reading, not compiled, not run on hardware
//
#include <hip/hip_runtime.h>

// ---------------------------------------------------------------------------
// AttentionFusion on MI455X (gfx1250, wave32, WMMA).
// attn = softmax(scale/64 * x·[sp;ms]^T);  out = W·(attn·[sp;ms]) + b
// Three bf16 WMMA GEMMs (f32 accumulate) + row softmax.
// Double-buffered LDS pipeline; async global->LDS copies where available.
// ---------------------------------------------------------------------------

#define DEV __device__ __forceinline__

typedef __bf16 v16bf __attribute__((ext_vector_type(16)));
typedef float  v8f   __attribute__((ext_vector_type(8)));

union FragAB { v16bf v; uint4 u[2]; };

constexpr int Bsz = 8;
constexpr int Cch = 512;
constexpr int Lsp = 64 * 64;     // 4096
constexpr int KVC = 2 * Cch;     // 1024

constexpr int BM = 128, BN = 128, BK = 32;
constexpr int LDSS = BK + 8;     // 40 bf16 = 80 B row stride (conflict-avoiding)

// ---- CDNA5 async global->LDS (ASYNCcnt-tracked DMA into LDS) --------------
#if defined(__has_builtin)
#  if __has_builtin(__builtin_amdgcn_global_load_async_to_lds_b128)
#    define ASYNC_LDS 1
#  endif
#endif
#ifndef ASYNC_LDS
#  define ASYNC_LDS 0
#endif

#if ASYNC_LDS
typedef int v4i __attribute__((vector_size(16)));
typedef __attribute__((address_space(1))) v4i* as1_v4i_p;   // global int4*
typedef __attribute__((address_space(3))) v4i* as3_v4i_p;   // LDS    int4*

DEV void async_cp16(const unsigned short* g, unsigned short* l) {
  // generic LDS pointer low 32 bits == LDS byte offset
  __builtin_amdgcn_global_load_async_to_lds_b128(
      (as1_v4i_p)(unsigned long long)(const void*)g,
      (as3_v4i_p)(unsigned int)(unsigned long long)(const void*)l,
      0, 0);
}
DEV void async_wait0() {
#  if __has_builtin(__builtin_amdgcn_s_wait_asynccnt)
  __builtin_amdgcn_s_wait_asynccnt(0);
#  else
  asm volatile("s_wait_asynccnt 0x0" ::: "memory");
#  endif
}
#endif

// native f32->bf16 (RNE): fptrunc lowers to v_cvt_*bf16_f32 on gfx1250
DEV unsigned short f2bf(float f) {
  union { __bf16 h; unsigned short u; } cv;
  cv.h = (__bf16)f;
  return cv.u;
}

// ---------------------------------------------------------------------------
// Pack kv = concat(spatial, multi_scale) -> bf16 [B, 2C, L]
// ---------------------------------------------------------------------------
__global__ void __launch_bounds__(256)
pack_kv_kernel(const float* __restrict__ sp, const float* __restrict__ ms,
               unsigned short* __restrict__ kv) {
  const long g = ((long)blockIdx.x * 256 + threadIdx.x) * 8;   // flat elem idx
  const int b   = (int)(g >> 22);               // KVC*L = 2^22
  const int rem = (int)(g & ((1 << 22) - 1));
  const int k   = rem >> 12;                    // L = 2^12
  const int l   = rem & (Lsp - 1);
  const float* src = (k < Cch) ? (sp + ((long)b * Cch + k) * Lsp + l)
                               : (ms + ((long)b * Cch + (k - Cch)) * Lsp + l);
  float4 f0 = *(const float4*)src;
  float4 f1 = *(const float4*)(src + 4);
  unsigned short o[8] = { f2bf(f0.x), f2bf(f0.y), f2bf(f0.z), f2bf(f0.w),
                          f2bf(f1.x), f2bf(f1.y), f2bf(f1.z), f2bf(f1.w) };
  *(uint4*)(kv + g) = *(const uint4*)o;
}

// ---------------------------------------------------------------------------
// Tiled WMMA GEMM: C[M,N] = alpha * A[M,K] * op(B) (+ bias)
//   A_F32 : A is fp32 in global (convert->bf16 while staging to LDS)
//   B_NT  : B is [N,K] row-major (k-contiguous)  else B is [K,N] row-major
//   256 threads = 8 waves; wave grid 2(M) x 4(N); wave tile 64x32
//   Double-buffered LDS; next tile issued before the WMMA burst.
// ---------------------------------------------------------------------------
template<bool A_F32, bool B_NT, bool OUT_BF16, bool HAS_BIAS>
__global__ void __launch_bounds__(256)
gemm_wmma_kernel(const void* __restrict__ Aptr,
                 const unsigned short* __restrict__ Bptr,
                 void* __restrict__ Cptr,
                 int M, int N, int K,
                 int lda, int ldb, int ldc,
                 long batchA, long batchB, long batchC,
                 const float* __restrict__ alpha_ptr, float alpha_const,
                 const float* __restrict__ bias) {
  __shared__ unsigned short lsA[2][BM][LDSS];
  __shared__ unsigned short lsB[2][BN][LDSS];

  const int bz   = blockIdx.z;
  const int m0   = blockIdx.y * BM;
  const int n0   = blockIdx.x * BN;
  const int tid  = threadIdx.x;
  const int lane = tid & 31;
  const int wid  = tid >> 5;
  const int lhalf = lane >> 4;     // 0/1: selects k-half per WMMA layout
  const int llo   = lane & 15;

  const float*          Af = (const float*)Aptr          + bz * batchA;
  const unsigned short* Ah = (const unsigned short*)Aptr + bz * batchA;
  const unsigned short* Bb = Bptr + bz * batchB;

  float alpha = alpha_const;
  if (alpha_ptr) alpha *= *alpha_ptr;

  const int wm = (wid & 1) * 64;   // wave M offset in block tile
  const int wn = (wid >> 1) * 32;  // wave N offset in block tile

  // copy-slice geometry (per thread)
  const int rowHalf = tid >> 1;          // 0..127, 2 threads/row, 16 elems each
  const int kcHalf  = (tid & 1) * 16;
  const int krNN    = tid >> 3;          // 0..31 (B[K,N] path)
  const int ncNN    = (tid & 7) * 16;    // 0..112

  const float*          aSrcF  = Af + (long)(m0 + rowHalf) * lda + kcHalf;
  const unsigned short* aSrcH  = Ah + (long)(m0 + rowHalf) * lda + kcHalf;
  const unsigned short* bSrcNT = Bb + (long)(n0 + rowHalf) * ldb + kcHalf;
  const unsigned short* bSrcNN = Bb + (long)krNN * ldb + n0 + ncNN;

  // staging registers (manual paths)
  float4 afr4[4];
  uint4  aH[2], bNT[2], bNN[2];

  auto ISSUE = [&](int kt, int nbuf) {
    const int k0 = kt * BK;
    if (A_F32) {
      const float* s = aSrcF + k0;
      afr4[0] = *(const float4*)(s);
      afr4[1] = *(const float4*)(s + 4);
      afr4[2] = *(const float4*)(s + 8);
      afr4[3] = *(const float4*)(s + 12);
    } else {
#if ASYNC_LDS
      async_cp16(aSrcH + k0,     &lsA[nbuf][rowHalf][kcHalf]);
      async_cp16(aSrcH + k0 + 8, &lsA[nbuf][rowHalf][kcHalf + 8]);
#else
      aH[0] = *(const uint4*)(aSrcH + k0);
      aH[1] = *(const uint4*)(aSrcH + k0 + 8);
#endif
    }
    if (B_NT) {
#if ASYNC_LDS
      async_cp16(bSrcNT + k0,     &lsB[nbuf][rowHalf][kcHalf]);
      async_cp16(bSrcNT + k0 + 8, &lsB[nbuf][rowHalf][kcHalf + 8]);
#else
      bNT[0] = *(const uint4*)(bSrcNT + k0);
      bNT[1] = *(const uint4*)(bSrcNT + k0 + 8);
#endif
    } else {
      const unsigned short* s = bSrcNN + (long)k0 * ldb;
      bNN[0] = *(const uint4*)s;
      bNN[1] = *(const uint4*)(s + 8);
    }
  };

  auto COMMIT = [&](int nbuf) {
    if (A_F32) {
      unsigned short o[16] = {
        f2bf(afr4[0].x), f2bf(afr4[0].y), f2bf(afr4[0].z), f2bf(afr4[0].w),
        f2bf(afr4[1].x), f2bf(afr4[1].y), f2bf(afr4[1].z), f2bf(afr4[1].w),
        f2bf(afr4[2].x), f2bf(afr4[2].y), f2bf(afr4[2].z), f2bf(afr4[2].w),
        f2bf(afr4[3].x), f2bf(afr4[3].y), f2bf(afr4[3].z), f2bf(afr4[3].w) };
      *(uint4*)&lsA[nbuf][rowHalf][kcHalf]     = *(const uint4*)&o[0];
      *(uint4*)&lsA[nbuf][rowHalf][kcHalf + 8] = *(const uint4*)&o[8];
    }
#if !ASYNC_LDS
    if (!A_F32) {
      *(uint4*)&lsA[nbuf][rowHalf][kcHalf]     = aH[0];
      *(uint4*)&lsA[nbuf][rowHalf][kcHalf + 8] = aH[1];
    }
    if (B_NT) {
      *(uint4*)&lsB[nbuf][rowHalf][kcHalf]     = bNT[0];
      *(uint4*)&lsB[nbuf][rowHalf][kcHalf + 8] = bNT[1];
    }
#endif
    if (!B_NT) {                      // transpose on store: lsB[n][k]
      unsigned short t[16];
      *(uint4*)&t[0] = bNN[0];
      *(uint4*)&t[8] = bNN[1];
#pragma unroll
      for (int j = 0; j < 16; ++j) lsB[nbuf][ncNN + j][krNN] = t[j];
    }
  };

  v8f acc[4][2];
#pragma unroll
  for (int i = 0; i < 4; ++i)
#pragma unroll
    for (int j = 0; j < 2; ++j) acc[i][j] = v8f{};

  const int kTiles = K / BK;

  // prologue: fill buffer 0
  ISSUE(0, 0);
  COMMIT(0);
#if ASYNC_LDS
  if (!A_F32 || B_NT) async_wait0();
#endif
  __syncthreads();

  for (int kt = 0; kt < kTiles; ++kt) {
    const int  buf  = kt & 1;
    const bool more = (kt + 1) < kTiles;

    if (more) ISSUE(kt + 1, buf ^ 1);   // globals / async DMA in flight

    // ---- fragments per documented gfx1250 16-bit WMMA layouts ----
    FragAB afr[4];
#pragma unroll
    for (int i = 0; i < 4; ++i) {
      const int r = wm + i * 16 + llo;        // A: lane&15 = row M
      afr[i].u[0] = *(const uint4*)&lsA[buf][r][lhalf * 8];      // K {0..7|8..15}
      afr[i].u[1] = *(const uint4*)&lsA[buf][r][16 + lhalf * 8]; // K {16..23|24..31}
    }
    FragAB bfr[2];
#pragma unroll
    for (int j = 0; j < 2; ++j) {
      const int c = wn + j * 16 + llo;        // B: lane&15 = col N
      bfr[j].u[0] = *(const uint4*)&lsB[buf][c][lhalf * 8];      // K {0..7|8..15}
      bfr[j].u[1] = *(const uint4*)&lsB[buf][c][16 + lhalf * 8]; // K {16..23|24..31}
    }

#pragma unroll
    for (int i = 0; i < 4; ++i)
#pragma unroll
      for (int j = 0; j < 2; ++j)
        acc[i][j] = __builtin_amdgcn_wmma_f32_16x16x32_bf16(
            false, afr[i].v, false, bfr[j].v, (short)0, acc[i][j], false, false);

    if (more) COMMIT(buf ^ 1);
#if ASYNC_LDS
    if (more && (!A_F32 || B_NT)) async_wait0();
#endif
    __syncthreads();
  }

  // ---- epilogue: C/D layout -> VGPR v holds M = v + (lane>=16)*8 ----
#pragma unroll
  for (int i = 0; i < 4; ++i) {
    const int mbase = m0 + wm + i * 16 + lhalf * 8;
#pragma unroll
    for (int j = 0; j < 2; ++j) {
      const int n = n0 + wn + j * 16 + llo;
#pragma unroll
      for (int v = 0; v < 8; ++v) {
        const int m = mbase + v;
        float val = acc[i][j][v] * alpha;
        if (HAS_BIAS) val += bias[m];
        if (OUT_BF16)
          ((unsigned short*)Cptr)[bz * batchC + (long)m * ldc + n] = f2bf(val);
        else
          ((float*)Cptr)[bz * batchC + (long)m * ldc + n] = val;
      }
    }
  }
}

// ---------------------------------------------------------------------------
// Row softmax over 1024 cols (f32 in -> normalized bf16 out). 1 block / row.
// ---------------------------------------------------------------------------
__global__ void __launch_bounds__(256)
softmax_rows_kernel(const float* __restrict__ attn, unsigned short* __restrict__ out) {
  __shared__ float red[256];
  const int row = blockIdx.x;
  const int tid = threadIdx.x;
  const float* src = attn + (long)row * KVC;

  float4 v = *(const float4*)(src + tid * 4);
  float mx = fmaxf(fmaxf(v.x, v.y), fmaxf(v.z, v.w));
  red[tid] = mx; __syncthreads();
#pragma unroll
  for (int s = 128; s > 0; s >>= 1) {
    if (tid < s) red[tid] = fmaxf(red[tid], red[tid + s]);
    __syncthreads();
  }
  mx = red[0]; __syncthreads();

  float e0 = __expf(v.x - mx), e1 = __expf(v.y - mx);
  float e2 = __expf(v.z - mx), e3 = __expf(v.w - mx);
  red[tid] = e0 + e1 + e2 + e3; __syncthreads();
#pragma unroll
  for (int s = 128; s > 0; s >>= 1) {
    if (tid < s) red[tid] += red[tid + s];
    __syncthreads();
  }
  const float inv = __frcp_rn(red[0]);

  unsigned p0 = (unsigned)f2bf(e0 * inv) | ((unsigned)f2bf(e1 * inv) << 16);
  unsigned p1 = (unsigned)f2bf(e2 * inv) | ((unsigned)f2bf(e3 * inv) << 16);
  ((uint2*)(out + (long)row * KVC))[tid] = make_uint2(p0, p1);
}

// ---------------------------------------------------------------------------
extern "C" void kernel_launch(void* const* d_in, const int* in_sizes, int n_in,
                              void* d_out, int out_size, void* d_ws, size_t ws_size,
                              hipStream_t stream) {
  const float* x       = (const float*)d_in[0];
  const float* spatial = (const float*)d_in[1];
  const float* multi   = (const float*)d_in[2];
  const float* scale   = (const float*)d_in[3];   // device scalar
  const float* conv_w  = (const float*)d_in[4];
  const float* conv_b  = (const float*)d_in[5];
  float*       out     = (float*)d_out;

  // workspace carve-up
  char* ws = (char*)d_ws;
  unsigned short* kv_bf16 = (unsigned short*)ws;                       // 64 MB
  size_t off = (size_t)Bsz * KVC * Lsp * 2;
  float* attn_f32 = (float*)(ws + off);                                // 16 MB
  off += (size_t)Bsz * Cch * KVC * 4;
  unsigned short* attn_bf16 = (unsigned short*)(ws + off);             // 8 MB
  off += (size_t)Bsz * Cch * KVC * 2;
  unsigned short* fused_bf16 = (unsigned short*)(ws + off);            // 32 MB

  // 1) kv = bf16(concat(spatial, multi))
  {
    const long nvec = (long)Bsz * KVC * Lsp / 8;
    pack_kv_kernel<<<dim3((unsigned)(nvec / 256)), dim3(256), 0, stream>>>(
        spatial, multi, kv_bf16);
  }

  // 2) attn_f32 = (scale/64) * x · kv^T    [M=512, N=1024, K=4096]  (NT)
  gemm_wmma_kernel<true, true, false, false>
      <<<dim3(KVC / BN, Cch / BM, Bsz), dim3(256), 0, stream>>>(
          x, kv_bf16, attn_f32, Cch, KVC, Lsp,
          Lsp, Lsp, KVC,
          (long)Cch * Lsp, (long)KVC * Lsp, (long)Cch * KVC,
          scale, 0.015625f, nullptr);

  // 3) softmax rows -> bf16
  softmax_rows_kernel<<<dim3(Bsz * Cch), dim3(256), 0, stream>>>(attn_f32, attn_bf16);

  // 4) fused_bf16 = attn · kv              [M=512, N=4096, K=1024]  (NN)
  gemm_wmma_kernel<false, false, true, false>
      <<<dim3(Lsp / BN, Cch / BM, Bsz), dim3(256), 0, stream>>>(
          attn_bf16, kv_bf16, fused_bf16, Cch, Lsp, KVC,
          KVC, Lsp, Lsp,
          (long)Cch * KVC, (long)KVC * Lsp, (long)Cch * Lsp,
          nullptr, 1.0f, nullptr);

  // 5) out = conv_w · fused + conv_b       [M=512, N=4096, K=512]   (NN)
  gemm_wmma_kernel<true, false, false, true>
      <<<dim3(Lsp / BN, Cch / BM, Bsz), dim3(256), 0, stream>>>(
          conv_w, fused_bf16, out, Cch, Lsp, Cch,
          Cch, Lsp, Lsp,
          0L, (long)Cch * Lsp, (long)Cch * Lsp,
          nullptr, 1.0f, conv_b);
}
